// InfoNCELoss_51737176047742
// MI455X (gfx1250) — matrix operand order repaired and checked by reading, not compiled, not run on hardware
//
#include <hip/hip_runtime.h>
#include <math.h>

// ---------------------------------------------------------------------------
// InfoNCE on MI455X (gfx1250, wave32):
//   cos_sim = (p1/||p1||) @ (p2/||p2||)^T  via split-f16 WMMA (f32-accurate)
//   loss    = mean_i( log(sum_j exp(2*cos[i,j])) - 2*cos[i,i] )
//
// Pipeline:
//   prep:  row norms -> normalized rows split into (hi,lo) f16 arrays in ws
//   gemm:  async global->LDS staging (ASYNCcnt), 3x f16 WMMA per K-step,
//          fused exp row-sum epilogue
//   loss:  single-block deterministic reduction
// ---------------------------------------------------------------------------

#define NROWS 8192
#define DDIM  128
#define LDAH  136   // padded LDS row stride in halves (272 B -> conflict-free)

typedef __attribute__((ext_vector_type(16))) _Float16 v16h;
typedef __attribute__((ext_vector_type(8)))  _Float16 v8h;
typedef __attribute__((ext_vector_type(4)))  _Float16 v4h;
typedef __attribute__((ext_vector_type(8)))  float    v8f;

union FragU { v16h v; v8h h[2]; };

// ---------------------------------------------------------------------------
// Kernel 1: per-row inverse norms (eps-clamped), write normalized rows as
// split f16 hi/lo arrays, zero the row-sum accumulator.
// 2048 blocks x 256 threads; one wave per row (32 lanes x float4 = 128 elems).
// ---------------------------------------------------------------------------
__global__ __launch_bounds__(256) void prep_kernel(
    const float* __restrict__ p1, const float* __restrict__ p2,
    _Float16* __restrict__ n1hi, _Float16* __restrict__ n1lo,
    _Float16* __restrict__ n2hi, _Float16* __restrict__ n2lo,
    float* __restrict__ rowsum) {
  const int lane = threadIdx.x & 31;
  const int wave = threadIdx.x >> 5;
  const bool isB = blockIdx.x >= 1024;
  const int  blk = isB ? (blockIdx.x - 1024) : blockIdx.x;
  const int  row = blk * 8 + wave;
  const float* src = isB ? p2 : p1;

  const float4 q = *(const float4*)(src + (size_t)row * DDIM + lane * 4);
  float ss = q.x * q.x + q.y * q.y + q.z * q.z + q.w * q.w;
  ss += __shfl_xor(ss, 1, 32);
  ss += __shfl_xor(ss, 2, 32);
  ss += __shfl_xor(ss, 4, 32);
  ss += __shfl_xor(ss, 8, 32);
  ss += __shfl_xor(ss, 16, 32);          // all lanes hold the row sum-of-squares
  const float inv = 1.0f / fmaxf(sqrtf(ss), 1e-8f);   // torch eps clamp

  const float v0 = q.x * inv, v1 = q.y * inv, v2 = q.z * inv, v3 = q.w * inv;
  const _Float16 h0 = (_Float16)v0, h1 = (_Float16)v1,
                 h2 = (_Float16)v2, h3 = (_Float16)v3;
  v4h hi = {h0, h1, h2, h3};
  v4h lo = {(_Float16)(v0 - (float)h0), (_Float16)(v1 - (float)h1),
            (_Float16)(v2 - (float)h2), (_Float16)(v3 - (float)h3)};
  const size_t off = (size_t)row * DDIM + lane * 4;
  *(v4h*)((isB ? n2hi : n1hi) + off) = hi;
  *(v4h*)((isB ? n2lo : n1lo) + off) = lo;

  if (!isB && lane == 0) rowsum[row] = 0.0f;
}

// ---------------------------------------------------------------------------
// Kernel 2: tiled GEMM + fused exp-row-sum epilogue.
// Grid 64x64, block 256 (8 waves). Block tile 128x128, wave tile 64x32.
// Staging uses gfx1250 async global->LDS copies (ASYNCcnt).
// ---------------------------------------------------------------------------
__global__ __launch_bounds__(256) void gemm_kernel(
    const _Float16* __restrict__ n1hi, const _Float16* __restrict__ n1lo,
    const _Float16* __restrict__ n2hi, const _Float16* __restrict__ n2lo,
    float* __restrict__ cos_out, float* __restrict__ rowsum) {
  extern __shared__ _Float16 smem[];
  _Float16* sAhi = smem;                 // arr 0
  _Float16* sAlo = sAhi + 128 * LDAH;    // arr 1
  _Float16* sBhi = sAlo + 128 * LDAH;    // arr 2
  _Float16* sBlo = sBhi + 128 * LDAH;    // arr 3

  const int tid   = threadIdx.x;
  const int lane  = tid & 31;
  const int wave  = tid >> 5;
  const int mBase = blockIdx.y * 128;
  const int nBase = blockIdx.x * 128;

  // ---- Phase 1: async-stage the four padded f16 tiles into LDS ------------
  // Per array: 128 rows x 16 b128 chunks/row = 2048 chunks; 8 iters x 256 thr.
  // LDS dst is padded (LDAH halves/row); global src is dense (DDIM halves).
  const uint32_t ldsBase = (uint32_t)(size_t)smem;
  const _Float16* const srcArr[4] = {n1hi, n1lo, n2hi, n2lo};
#pragma unroll
  for (int arr = 0; arr < 4; ++arr) {
    const _Float16* const src = srcArr[arr];
    const int rBase = (arr < 2) ? mBase : nBase;
    const uint32_t ldsArr = ldsBase + (uint32_t)(arr * 128 * LDAH * 2);
#pragma unroll
    for (int i = 0; i < 8; ++i) {
      const int c   = i * 256 + tid;   // 0..2047
      const int row = c >> 4;          // 0..127
      const int c16 = c & 15;          // 16-byte chunk within row
      const unsigned long long g =
          (unsigned long long)(size_t)(src + (size_t)(rBase + row) * DDIM + c16 * 8);
      const uint32_t l = ldsArr + (uint32_t)(row * (LDAH * 2) + c16 * 16);
      asm volatile("global_load_async_to_lds_b128 %0, %1, off"
                   :: "v"(l), "v"(g) : "memory");
    }
  }
  asm volatile("s_wait_asynccnt 0" ::: "memory");
  __syncthreads();

  // ---- Phase 2: WMMA main loop --------------------------------------------
  const int wm   = (wave >> 2) * 64;   // wave row offset in block tile
  const int wn   = (wave & 3) * 32;    // wave col offset in block tile
  const int l15  = lane & 15;
  const int hsel = lane >> 4;          // 0 for lanes 0-15, 1 for lanes 16-31

  v8f acc[4][2];
#pragma unroll
  for (int mi = 0; mi < 4; ++mi)
#pragma unroll
    for (int ni = 0; ni < 2; ++ni)
      acc[mi][ni] = (v8f){0.f, 0.f, 0.f, 0.f, 0.f, 0.f, 0.f, 0.f};

  for (int ks = 0; ks < 4; ++ks) {          // K = 4 x 32
    FragU ah[4], al[4], bh[2], bl[2];
#pragma unroll
    for (int mi = 0; mi < 4; ++mi) {
      // A layout (16-bit, 16x32): lanes 0-15 hold K=[0..7]+[16..23], lanes
      // 16-31 hold K=[8..15]+[24..31] for their row.
      const int off = (wm + mi * 16 + l15) * LDAH + ks * 32 + hsel * 8;
      ah[mi].h[0] = *(const v8h*)(sAhi + off);
      ah[mi].h[1] = *(const v8h*)(sAhi + off + 16);
      al[mi].h[0] = *(const v8h*)(sAlo + off);
      al[mi].h[1] = *(const v8h*)(sAlo + off + 16);
    }
#pragma unroll
    for (int ni = 0; ni < 2; ++ni) {
      // B layout (16-bit, 32x16): lane = column; lanes 0-15 hold K=[0..7]+[16..23],
      // lanes 16-31 hold K=[8..15]+[24..31] (same K order as A).
      // B column n == row n of n2 (B = n2^T), so rows stream along K.
      const int off = (wn + ni * 16 + l15) * LDAH + ks * 32 + hsel * 8;
      bh[ni].h[0] = *(const v8h*)(sBhi + off);
      bh[ni].h[1] = *(const v8h*)(sBhi + off + 16);
      bl[ni].h[0] = *(const v8h*)(sBlo + off);
      bl[ni].h[1] = *(const v8h*)(sBlo + off + 16);
    }
#pragma unroll
    for (int mi = 0; mi < 4; ++mi) {
#pragma unroll
      for (int ni = 0; ni < 2; ++ni) {
        acc[mi][ni] = __builtin_amdgcn_wmma_f32_16x16x32_f16(
            false, ah[mi].v, false, bh[ni].v, (short)0, acc[mi][ni], false, false);
        acc[mi][ni] = __builtin_amdgcn_wmma_f32_16x16x32_f16(
            false, ah[mi].v, false, bl[ni].v, (short)0, acc[mi][ni], false, false);
        acc[mi][ni] = __builtin_amdgcn_wmma_f32_16x16x32_f16(
            false, al[mi].v, false, bh[ni].v, (short)0, acc[mi][ni], false, false);
      }
    }
  }

  // ---- Phase 3: epilogue — store cos_sim, fused exp row-sums --------------
  // C/D layout: VGPR r -> M = r + hsel*8, N = l15.
#pragma unroll
  for (int mi = 0; mi < 4; ++mi) {
#pragma unroll
    for (int r = 0; r < 8; ++r) {
      const int gm = mBase + wm + mi * 16 + hsel * 8 + r;
      const float v0 = acc[mi][0][r];
      const float v1 = acc[mi][1][r];
      float* dst = cos_out + (size_t)gm * NROWS + nBase + wn + l15;
      dst[0]  = v0;
      dst[16] = v1;
      // exp(cos/T), T = 0.5; values in [-2,2] so no max-shift needed.
      float e = __expf(2.0f * v0) + __expf(2.0f * v1);
      e += __shfl_xor(e, 1, 32);
      e += __shfl_xor(e, 2, 32);
      e += __shfl_xor(e, 4, 32);
      e += __shfl_xor(e, 8, 32);
      if (l15 == 0) atomicAdd(&rowsum[gm], e);
    }
  }
}

// ---------------------------------------------------------------------------
// Kernel 3: loss = mean_i( log(rowsum[i]) - 2*cos[i,i] ).  Single block,
// fully deterministic reduction, direct write (no pre-zero needed).
// ---------------------------------------------------------------------------
__global__ __launch_bounds__(256) void loss_kernel(
    const float* __restrict__ cos_out, const float* __restrict__ rowsum,
    float* __restrict__ loss) {
  __shared__ float red[256];
  float s = 0.0f;
  for (int i = threadIdx.x; i < NROWS; i += 256) {
    s += __logf(rowsum[i]) - 2.0f * cos_out[(size_t)i * NROWS + i];
  }
  red[threadIdx.x] = s;
  __syncthreads();
  for (int off = 128; off > 0; off >>= 1) {
    if (threadIdx.x < off) red[threadIdx.x] += red[threadIdx.x + off];
    __syncthreads();
  }
  if (threadIdx.x == 0) *loss = red[0] * (1.0f / NROWS);
}

// ---------------------------------------------------------------------------
extern "C" void kernel_launch(void* const* d_in, const int* in_sizes, int n_in,
                              void* d_out, int out_size, void* d_ws, size_t ws_size,
                              hipStream_t stream) {
  const float* p1 = (const float*)d_in[0];
  const float* p2 = (const float*)d_in[1];
  float* out = (float*)d_out;                 // [N*N] cos_sim, then [1] loss

  _Float16* n1hi = (_Float16*)d_ws;                    // 2 MB each
  _Float16* n1lo = n1hi + (size_t)NROWS * DDIM;
  _Float16* n2hi = n1lo + (size_t)NROWS * DDIM;
  _Float16* n2lo = n2hi + (size_t)NROWS * DDIM;
  float*  rowsum = (float*)(n2lo + (size_t)NROWS * DDIM);  // 32 KB

  prep_kernel<<<2048, 256, 0, stream>>>(p1, p2, n1hi, n1lo, n2hi, n2lo, rowsum);

  const size_t smem_bytes = 4u * 128u * LDAH * sizeof(_Float16);  // ~136 KB
  dim3 grid(NROWS / 128, NROWS / 128);
  gemm_kernel<<<grid, 256, smem_bytes, stream>>>(n1hi, n1lo, n2hi, n2lo, out, rowsum);

  loss_kernel<<<1, 256, 0, stream>>>(out, rowsum, out + (size_t)NROWS * NROWS);
}
